// SymbolicOperator_49495203119466
// MI455X (gfx1250) — matrix operand order repaired and checked by reading, not compiled, not run on hardware
//
#include <hip/hip_runtime.h>

typedef __attribute__((ext_vector_type(16))) __bf16 v16bf;
typedef __attribute__((ext_vector_type(8)))  __bf16 v8bf;
typedef __attribute__((ext_vector_type(8)))  float  v8f;
typedef __attribute__((ext_vector_type(4)))  float  v4f;

#define BATCH     2048
#define S_WORDS   8
#define NSTEPS    3
#define L_LEN     50
#define LPAD      64
#define K_DIM     128
#define OUT_V     128
#define H_DIM     384
#define P_DIM     200
#define P_PAD     224
#define BT        16

#define N_WIH (3*H_DIM*P_PAD)   /* 1152*224 bf16 */
#define N_WHH (3*H_DIM*H_DIM)   /* 1152*384 bf16 */
#define N_KEY (LPAD*K_DIM)      /* 64*128  bf16 */

__device__ __forceinline__ unsigned short f2bf(float x){
  unsigned u = __float_as_uint(x);
  unsigned r = ((u >> 16) & 1u) + 0x7FFFu;     // round-to-nearest-even
  return (unsigned short)((u + r) >> 16);
}
__device__ __forceinline__ float bf2f(unsigned short b){
  return __uint_as_float(((unsigned)b) << 16);
}
__device__ __forceinline__ float sigm(float x){ return 1.f/(1.f + __expf(-x)); }

__device__ __forceinline__ v8f wmma_bf16(v16bf a, v16bf b, v8f c){
  return __builtin_amdgcn_wmma_f32_16x16x32_bf16(false, a, false, b, (short)0, c, false, false);
}

// A-operand frag (16x32 bf16, M=lane&15): lane<16 holds K k0..k0+15,
// lane>=16 holds K k0+16..k0+31.
__device__ __forceinline__ v16bf a_frag_lds(const unsigned short* base, int rowStride,
                                            int m, int k0, int sel){
  const unsigned short* p = base + m*rowStride + k0 + sel*16;
  v8bf lo = *(const v8bf*)p;          // 16B ds load
  v8bf hi = *(const v8bf*)(p + 8);    // 16B ds load
  return __builtin_shufflevector(lo, hi, 0,1,2,3,4,5,6,7,8,9,10,11,12,13,14,15);
}

// B-operand frag (32x16 bf16, N=lane&15): lane<16 -> K k0..k0+15, lane>=16 -> k0+16..k0+31.
// Row-major weight rows W[n][k] feed this directly (no transpose).
__device__ __forceinline__ v16bf b_frag_g(const unsigned short* w, int row, int rowStride,
                                          int k0, int sel){
  return *(const v16bf*)(w + (size_t)row*rowStride + k0 + sel*16);
}

__global__ void prep_kernel(const float* __restrict__ Wih, const float* __restrict__ Whh,
                            const float* __restrict__ keys, unsigned short* __restrict__ ws){
  int idx = blockIdx.x * blockDim.x + threadIdx.x;
  if (idx < N_WIH){
    int row = idx / P_PAD, k = idx - row*P_PAD;
    float v = (k < P_DIM) ? Wih[row*P_DIM + k] : 0.f;
    ws[idx] = f2bf(v);
  } else if (idx < N_WIH + N_WHH){
    int j = idx - N_WIH;
    ws[idx] = f2bf(Whh[j]);
  } else if (idx < N_WIH + N_WHH + N_KEY){
    int j = idx - N_WIH - N_WHH;
    int row = j >> 7;
    float v = (row < L_LEN) ? keys[j] : 0.f;
    ws[idx] = f2bf(v);
  }
}

__global__ __launch_bounds__(256) void sym_op_kernel(
    const int*   __restrict__ inst,     const float* __restrict__ gate_emb,
    const float* __restrict__ prog_emb, const float* __restrict__ prim_emb,
    const float* __restrict__ b_ih,     const float* __restrict__ b_hh,
    const float* __restrict__ keys_f,   const float* __restrict__ init_sv,
    const unsigned short* __restrict__ ws, float* __restrict__ out)
{
  __shared__ __align__(16) float          h_lds[BT*H_DIM];     // 24.6 KB
  __shared__ __align__(16) unsigned short hb_lds[BT*H_DIM];    // 12.3 KB
  __shared__ __align__(16) unsigned short prog_lds[BT*P_PAD];  //  7.2 KB
  __shared__ __align__(16) unsigned short prim_lds[BT*OUT_V];  //  4.1 KB
  __shared__               float          gate_lds[BT*2];
  __shared__ __align__(16) float          rattn[BT*LPAD];      //  4.1 KB
  __shared__ __align__(16) float          wmask[BT*LPAD];      //  4.1 KB
  __shared__               int            words[BT];

  const int tid  = threadIdx.x;
  const int lane = tid & 31;
  const int wave = tid >> 5;
  const int m0   = lane & 15;
  const int sel  = lane >> 4;
  const int b0   = blockIdx.x * BT;

  const unsigned short* Wih_b  = ws;
  const unsigned short* Whh_b  = ws + N_WIH;
  const unsigned short* keys_b = ws + N_WIH + N_WHH;

  // h0[b][j] = keys[0][j % 128] ; 384 % 128 == 0 so idx&127 indexes the key column
  for (int idx = tid; idx < BT*H_DIM; idx += 256){
    float x = keys_f[idx & 127];
    h_lds[idx]  = x;
    hb_lds[idx] = f2bf(x);
  }
  // sv0 broadcast of init_scratch_value into d_out (in-place state, L2-resident)
  {
    v4f*       out4 = (v4f*)out + (size_t)b0*(L_LEN*OUT_V/4);
    const v4f* iv4  = (const v4f*)init_sv;
    for (int idx = tid; idx < BT*L_LEN*OUT_V/4; idx += 256)
      out4[idx] = iv4[idx & 31];                 // 128-float period = 32 float4s
  }
  __syncthreads();

  for (int w = 0; w < S_WORDS; ++w){
    // ---- word setup: word id, gate softmax(2), prog (bf16, K-padded), prim (bf16) ----
    if (tid < BT){
      int wd = inst[w*BATCH + b0 + tid];
      words[tid] = wd;
      float g0 = gate_emb[wd*2+0], g1 = gate_emb[wd*2+1];
      float mx = fmaxf(g0, g1);
      float e0 = __expf(g0 - mx), e1 = __expf(g1 - mx);
      float is = 1.f/(e0 + e1);
      gate_lds[tid*2+0] = e0*is;
      gate_lds[tid*2+1] = e1*is;
    }
    __syncthreads();
    for (int idx = tid; idx < BT*P_PAD; idx += 256){
      int i = idx / P_PAD, k = idx - i*P_PAD;
      float v = (k < P_DIM) ? prog_emb[words[i]*P_DIM + k] : 0.f;
      prog_lds[idx] = f2bf(v);
    }
    for (int idx = tid; idx < BT*OUT_V; idx += 256){
      int i = idx >> 7, v = idx & 127;
      prim_lds[idx] = f2bf(prim_emb[words[i]*OUT_V + v]);
    }
    __syncthreads();

    for (int step = 0; step < NSTEPS; ++step){
      // ---- phase 1: attention logits via WMMA (waves 0-3: read ptr, 4-7: write ptr) ----
      {
        int    nt   = wave & 3;
        int    hoff = (wave < 4) ? 0 : K_DIM;
        float* obuf = (wave < 4) ? rattn : wmask;
        v8f acc = {};
        v16bf a = a_frag_lds(hb_lds + hoff, H_DIM, m0, 0, sel);
        v16bf b = b_frag_g(keys_b, nt*16 + m0, K_DIM, 0, sel);
        #pragma unroll 1
        for (int k0 = 0; k0 < K_DIM - 32; k0 += 32){
          v16bf a2 = a_frag_lds(hb_lds + hoff, H_DIM, m0, k0 + 32, sel);
          v16bf b2 = b_frag_g(keys_b, nt*16 + m0, K_DIM, k0 + 32, sel);
          acc = wmma_bf16(a, b, acc);
          a = a2; b = b2;
        }
        acc = wmma_bf16(a, b, acc);
        #pragma unroll
        for (int v = 0; v < 8; ++v)
          obuf[(v + 8*sel)*LPAD + nt*16 + m0] = acc[v];
      }
      __syncthreads();
      // ---- phase 2: masked softmax rows (L=50; padded cols never read) ----
      if (tid < 32){
        float* buf = ((tid >> 4) ? wmask : rattn) + (tid & 15)*LPAD;
        float mx = -3.4e38f;
        for (int l = 0; l < L_LEN; ++l) mx = fmaxf(mx, buf[l]);
        float s = 0.f;
        for (int l = 0; l < L_LEN; ++l){ float e = __expf(buf[l] - mx); buf[l] = e; s += e; }
        float inv = 1.f/s;
        for (int l = 0; l < L_LEN; ++l) buf[l] *= inv;
      }
      __syncthreads();
      // ---- phase 3+4: read_value, new_value, sv blend-update (float4 stream via L2) ----
      {
        int vg = tid & 31;                     // float4 column group (32 x 4 = 128 cols)
        #pragma unroll
        for (int rep = 0; rep < 2; ++rep){
          int i = (tid >> 5) + 8*rep;
          v4f* svp = (v4f*)(out + (size_t)(b0 + i)*(L_LEN*OUT_V)) + vg;
          const float* ra = rattn + i*LPAD;
          v4f rv = {0.f, 0.f, 0.f, 0.f};
          #pragma unroll 5
          for (int l = 0; l < L_LEN; ++l) rv += ra[l] * svp[l*(OUT_V/4)];
          const unsigned short* pp = prim_lds + i*OUT_V + vg*4;
          v4f pm = { bf2f(pp[0]), bf2f(pp[1]), bf2f(pp[2]), bf2f(pp[3]) };
          float g0 = gate_lds[i*2+0], g1 = gate_lds[i*2+1];
          v4f nv = g0*pm + g1*rv;
          const float* wm = wmask + i*LPAD;
          #pragma unroll 5
          for (int l = 0; l < L_LEN; ++l){
            v4f s = svp[l*(OUT_V/4)];
            svp[l*(OUT_V/4)] = s + wm[l]*(nv - s);   // wm*nv + (1-wm)*s
          }
        }
      }
      // ---- phase 5: GRU. gi + gh per 16-col block via WMMA; r/z/n fused in epilogue ----
      v8f hnew[3];
      #pragma unroll
      for (int t = 0; t < 3; ++t){
        int n0 = (wave + 8*t) * 16;
        v8f ir = {}, iz = {}, in_ = {};
        {
          v16bf a  = a_frag_lds(prog_lds, P_PAD, m0, 0, sel);
          v16bf br = b_frag_g(Wih_b,           n0 + m0, P_PAD, 0, sel);
          v16bf bz = b_frag_g(Wih_b,   H_DIM + n0 + m0, P_PAD, 0, sel);
          v16bf bn = b_frag_g(Wih_b, 2*H_DIM + n0 + m0, P_PAD, 0, sel);
          #pragma unroll 1
          for (int k0 = 0; k0 < P_PAD - 32; k0 += 32){
            v16bf a2  = a_frag_lds(prog_lds, P_PAD, m0, k0 + 32, sel);
            v16bf br2 = b_frag_g(Wih_b,           n0 + m0, P_PAD, k0 + 32, sel);
            v16bf bz2 = b_frag_g(Wih_b,   H_DIM + n0 + m0, P_PAD, k0 + 32, sel);
            v16bf bn2 = b_frag_g(Wih_b, 2*H_DIM + n0 + m0, P_PAD, k0 + 32, sel);
            ir  = wmma_bf16(a, br, ir);
            iz  = wmma_bf16(a, bz, iz);
            in_ = wmma_bf16(a, bn, in_);
            a = a2; br = br2; bz = bz2; bn = bn2;
          }
          ir  = wmma_bf16(a, br, ir);
          iz  = wmma_bf16(a, bz, iz);
          in_ = wmma_bf16(a, bn, in_);
        }
        v8f hr = {}, hz = {}, hn = {};
        {
          v16bf a  = a_frag_lds(hb_lds, H_DIM, m0, 0, sel);
          v16bf br = b_frag_g(Whh_b,           n0 + m0, H_DIM, 0, sel);
          v16bf bz = b_frag_g(Whh_b,   H_DIM + n0 + m0, H_DIM, 0, sel);
          v16bf bn = b_frag_g(Whh_b, 2*H_DIM + n0 + m0, H_DIM, 0, sel);
          #pragma unroll 1
          for (int k0 = 0; k0 < H_DIM - 32; k0 += 32){
            v16bf a2  = a_frag_lds(hb_lds, H_DIM, m0, k0 + 32, sel);
            v16bf br2 = b_frag_g(Whh_b,           n0 + m0, H_DIM, k0 + 32, sel);
            v16bf bz2 = b_frag_g(Whh_b,   H_DIM + n0 + m0, H_DIM, k0 + 32, sel);
            v16bf bn2 = b_frag_g(Whh_b, 2*H_DIM + n0 + m0, H_DIM, k0 + 32, sel);
            hr = wmma_bf16(a, br, hr);
            hz = wmma_bf16(a, bz, hz);
            hn = wmma_bf16(a, bn, hn);
            a = a2; br = br2; bz = bz2; bn = bn2;
          }
          hr = wmma_bf16(a, br, hr);
          hz = wmma_bf16(a, bz, hz);
          hn = wmma_bf16(a, bn, hn);
        }
        int j = n0 + m0;
        float bir = b_ih[j],           bhr = b_hh[j];
        float biz = b_ih[H_DIM + j],   bhz = b_hh[H_DIM + j];
        float bin = b_ih[2*H_DIM + j], bhn = b_hh[2*H_DIM + j];
        #pragma unroll
        for (int v = 0; v < 8; ++v){
          float r  = sigm(ir[v]  + bir + hr[v] + bhr);
          float z  = sigm(iz[v]  + biz + hz[v] + bhz);
          float n  = tanhf(in_[v] + bin + r*(hn[v] + bhn));
          float ho = h_lds[(v + 8*sel)*H_DIM + j];
          hnew[t][v] = (1.f - z)*n + z*ho;
        }
      }
      __syncthreads();   // all waves done reading h/hb (k-loops & h_old)
      #pragma unroll
      for (int t = 0; t < 3; ++t){
        int j = (wave + 8*t)*16 + m0;
        #pragma unroll
        for (int v = 0; v < 8; ++v){
          int idx = (v + 8*sel)*H_DIM + j;
          float x = hnew[t][v];
          h_lds[idx]  = x;
          hb_lds[idx] = f2bf(x);
        }
      }
      __syncthreads();
    }
  }
}

extern "C" void kernel_launch(void* const* d_in, const int* in_sizes, int n_in,
                              void* d_out, int out_size, void* d_ws, size_t ws_size,
                              hipStream_t stream){
  (void)in_sizes; (void)n_in; (void)out_size; (void)ws_size;
  const int*   inst     = (const int*)  d_in[0];
  /* d_in[1] true_actions unused by the reference */
  const float* gate_emb = (const float*)d_in[2];
  const float* prog_emb = (const float*)d_in[3];
  const float* prim_emb = (const float*)d_in[4];
  const float* W_ih     = (const float*)d_in[5];
  const float* W_hh     = (const float*)d_in[6];
  const float* b_ih     = (const float*)d_in[7];
  const float* b_hh     = (const float*)d_in[8];
  const float* keys     = (const float*)d_in[9];
  const float* init_sv  = (const float*)d_in[10];
  float* out = (float*)d_out;
  unsigned short* ws = (unsigned short*)d_ws;

  int tot = N_WIH + N_WHH + N_KEY;
  prep_kernel<<<(tot + 255)/256, 256, 0, stream>>>(W_ih, W_hh, keys, ws);
  sym_op_kernel<<<BATCH/BT, 256, 0, stream>>>(inst, gate_emb, prog_emb, prim_emb,
                                              b_ih, b_hh, keys, init_sv, ws, out);
}
